// NvidiaMambaSSM_41248865911021
// MI455X (gfx1250) — matrix operand order repaired and checked by reading, not compiled, not run on hardware
//
#include <hip/hip_runtime.h>
#include <hip/hip_bf16.h>

// ---------------------------------------------------------------------------
// Mamba-like SSM forward for MI455X (gfx1250, wave32, WMMA).
// Pipeline: f32->bf16 cvt -> WMMA GEMM (in_proj, split u/res) -> depthwise
// conv -> WMMA GEMM (dt_proj, fused softplus/sigmoid gating) -> chunked decay
// scan -> WMMA GEMM (out_proj, fused +res).
// GEMMs: 128x128 block tile, LDS-staged via GLOBAL_LOAD_ASYNC_TO_LDS_B128
// (ASYNCcnt) when available, double-buffered over K, 8 waves x (64x32) wave
// tiles, bf16 WMMA / f32 accumulate.
// x_proj / A_log / D are dead code in the reference and skipped.
// ---------------------------------------------------------------------------

typedef __attribute__((ext_vector_type(16))) __bf16 v16bf;
typedef __attribute__((ext_vector_type(8)))  float  v8f;
typedef __attribute__((ext_vector_type(4)))  int    v4i;

#define D_MODEL 1024
#define D_CONV  4
#define D_INNER 2048
#define BATCH   4
#define SEQ     4096
#define MTOT    (BATCH * SEQ)          // 16384 rows

#define BK        32                   // K per stage (one WMMA K)
#define LDS_ROW   80                   // 64B data + 16B pad -> conflict-free ds_load_b128
#define TILE_BYTES (128 * LDS_ROW)     // 10 KB per 128x32 bf16 tile

#if defined(__gfx1250__) && \
    __has_builtin(__builtin_amdgcn_global_load_async_to_lds_b128) && \
    __has_builtin(__builtin_amdgcn_s_wait_asynccnt)
#define HAVE_ASYNC 1
#else
#define HAVE_ASYNC 0
#endif

union Frag {
    v16bf v;
    uint4 q[2];
};

// 16B copy global -> LDS (async direct-to-LDS on gfx1250).
__device__ __forceinline__ void cp16(void* lptr, const void* gptr) {
#if HAVE_ASYNC
    __builtin_amdgcn_global_load_async_to_lds_b128(
        (__attribute__((address_space(1))) v4i*)gptr,
        (__attribute__((address_space(3))) v4i*)lptr, 0, 0);
#else
    *(uint4*)lptr = *(const uint4*)gptr;
#endif
}

__device__ __forceinline__ void async_wait4() {
#if HAVE_ASYNC
    __builtin_amdgcn_s_wait_asynccnt(4);
#endif
}
__device__ __forceinline__ void async_wait0() {
#if HAVE_ASYNC
    __builtin_amdgcn_s_wait_asynccnt(0);
#endif
}

// Stage one 128x32 bf16 tile (rows grow0..grow0+127, K cols k0..k0+31) into LDS.
// 512 x 16B chunks, 2 per thread.
__device__ __forceinline__ void stage_tile(const __bf16* __restrict__ G, int grow0,
                                           int K, int k0,
                                           unsigned char* lbase, int tid) {
#pragma unroll
    for (int j = 0; j < 2; ++j) {
        const int c   = tid + j * 256;
        const int row = c >> 2;
        const int off = (c & 3) * 16;
        const char* g = (const char*)(G + (size_t)(grow0 + row) * K + k0) + off;
        cp16(lbase + row * LDS_ROW + off, g);
    }
}

// A fragment (16x32 bf16, ISA 7.12.2): lane r of half h -> row r,
// K bytes [16h,16h+16) and [32+16h,48+16h).
__device__ __forceinline__ v16bf lds_a_frag(const unsigned char* lbase, int row, int half) {
    Frag f;
    const uint4* p = (const uint4*)(lbase + row * LDS_ROW + half * 16);
    f.q[0] = p[0];
    f.q[1] = p[2];          // +32 bytes
    return f.v;
}

// B fragment (32x16, B(k,n)=W[n][k]): lane r holds W-row r, K bytes [16h,16h+16) and [32+16h,48+16h).
__device__ __forceinline__ v16bf lds_b_frag(const unsigned char* lbase, int row, int half) {
    Frag f;
    const uint4* p = (const uint4*)(lbase + row * LDS_ROW + half * 16);
    f.q[0] = p[0];
    f.q[1] = p[2];
    return f.v;
}

// ---------------------------------------------------------------------------
// WMMA GEMM: C[m,n] = sum_k A[m,k] * W[n,k]   (A: MxK bf16, W: NxK bf16)
// MODE 0: in_proj  -> split store: n<2048 -> u(f32), else -> res(f32)
// MODE 1: dt_proj  -> gated = ucb * sigmoid(softplus(acc + bias[n]))
// MODE 2: out_proj -> out = acc + res
// ---------------------------------------------------------------------------
template <int MODE>
__global__ __launch_bounds__(256) void gemm_wmma(
    const __bf16* __restrict__ A, const __bf16* __restrict__ W,
    int M, int N, int K,
    float* __restrict__ out0, float* __restrict__ out1,
    const float* __restrict__ bias, const __bf16* __restrict__ ucb)
{
    __shared__ __align__(16) unsigned char sA[2][TILE_BYTES];
    __shared__ __align__(16) unsigned char sB[2][TILE_BYTES];

    const int tid  = threadIdx.x;
    const int lane = tid & 31;
    const int wave = tid >> 5;
    const int half = (lane >> 4) & 1;
    const int r    = lane & 15;
    const int wm   = wave & 1;          // 2 wave rows  x 64
    const int wn   = wave >> 1;         // 4 wave cols  x 32

    const int gm0 = blockIdx.y * 128;   // block M origin
    const int gn0 = blockIdx.x * 128;   // block N origin
    const int m0  = gm0 + wm * 64;
    const int n0  = gn0 + wn * 32;

    v8f acc[4][2];
#pragma unroll
    for (int mt = 0; mt < 4; ++mt)
#pragma unroll
        for (int nt = 0; nt < 2; ++nt)
            acc[mt][nt] = (v8f){0.f, 0.f, 0.f, 0.f, 0.f, 0.f, 0.f, 0.f};

    const int NK = K / BK;

    // prologue: stage k-step 0 into buffer 0
    stage_tile(A, gm0, K, 0, sA[0], tid);
    stage_tile(W, gn0, K, 0, sB[0], tid);

    for (int ks = 0; ks < NK; ++ks) {
        const int cur = ks & 1;
        if (ks + 1 < NK) {
            stage_tile(A, gm0, K, (ks + 1) * BK, sA[cur ^ 1], tid);
            stage_tile(W, gn0, K, (ks + 1) * BK, sB[cur ^ 1], tid);
            async_wait4();              // current stage's 4 copies done; next 4 in flight
        } else {
            async_wait0();
        }
        __syncthreads();                // all waves' fills of buffer `cur` visible

        v16bf a[4], b[2];
#pragma unroll
        for (int mt = 0; mt < 4; ++mt)
            a[mt] = lds_a_frag(sA[cur], wm * 64 + mt * 16 + r, half);
#pragma unroll
        for (int nt = 0; nt < 2; ++nt)
            b[nt] = lds_b_frag(sB[cur], wn * 32 + nt * 16 + r, half);
#pragma unroll
        for (int mt = 0; mt < 4; ++mt)
#pragma unroll
            for (int nt = 0; nt < 2; ++nt)
                acc[mt][nt] = __builtin_amdgcn_wmma_f32_16x16x32_bf16(
                    false, a[mt], false, b[nt], (short)0, acc[mt][nt], false, false);

        __syncthreads();                // buffer `cur` free for the ks+1 prefetch
    }

    // C/D layout: VGPR i -> (M = i + 8*half, N = r)
#pragma unroll
    for (int mt = 0; mt < 4; ++mt) {
#pragma unroll
        for (int nt = 0; nt < 2; ++nt) {
            const int rowb = m0 + mt * 16 + half * 8;
            const int col  = n0 + nt * 16 + r;
            if (MODE == 0) {
                // u/res split is uniform per 16-wide N tile (2048 % 16 == 0)
                if (n0 + nt * 16 < D_INNER) {
#pragma unroll
                    for (int i = 0; i < 8; ++i)
                        out0[(size_t)(rowb + i) * D_INNER + col] = acc[mt][nt][i];
                } else {
#pragma unroll
                    for (int i = 0; i < 8; ++i)
                        out1[(size_t)(rowb + i) * D_MODEL + (col - D_INNER)] = acc[mt][nt][i];
                }
            } else if (MODE == 1) {
                const float bv = bias[col];
#pragma unroll
                for (int i = 0; i < 8; ++i) {
                    const float d  = acc[mt][nt][i] + bv;
                    const float sp = (d > 20.f) ? d : log1pf(__expf(d));
                    const float sg = 1.f / (1.f + __expf(-sp));
                    const float uv = (float)ucb[(size_t)(rowb + i) * D_INNER + col];
                    out0[(size_t)(rowb + i) * D_INNER + col] = uv * sg;
                }
            } else {
#pragma unroll
                for (int i = 0; i < 8; ++i)
                    out0[(size_t)(rowb + i) * D_MODEL + col] =
                        acc[mt][nt][i] + out1[(size_t)(rowb + i) * D_MODEL + col];
            }
        }
    }
}

// ---------------------------------------------------------------------------
__global__ __launch_bounds__(256) void f32_to_bf16(const float* __restrict__ src,
                                                   __bf16* __restrict__ dst,
                                                   size_t n) {
    size_t i = (size_t)blockIdx.x * blockDim.x + threadIdx.x;
    if (i < n) dst[i] = (__bf16)src[i];
}

// Depthwise causal conv (k=4) over sequence, + bias; output to bf16.
__global__ __launch_bounds__(256) void dwconv_kernel(const float* __restrict__ u,
                                                     const float* __restrict__ cw,
                                                     const float* __restrict__ cb,
                                                     __bf16* __restrict__ ucb) {
    size_t idx = (size_t)blockIdx.x * blockDim.x + threadIdx.x;  // over B*S*D_INNER
    const int c = (int)(idx % D_INNER);
    const int s = (int)((idx / D_INNER) % SEQ);
    const int b = (int)(idx / ((size_t)D_INNER * SEQ));
    float accv = cb[c];
#pragma unroll
    for (int k = 0; k < D_CONV; ++k) {
        const int ss = s - (D_CONV - 1) + k;
        if (ss >= 0)
            accv += u[((size_t)b * SEQ + ss) * D_INNER + c] * cw[c * D_CONV + k];
    }
    ucb[idx] = (__bf16)accv;
}

// Chunked EMA scan: state = 0.9*state + 0.1*g.  0.9^256 ~ 2e-12 << fp32 eps,
// so each 512-chunk warms up with a 256-step lookback -> 8x parallelism.
#define CHUNK 512
#define LOOK  256
__global__ __launch_bounds__(256) void scan_kernel(const float* __restrict__ g,
                                                   __bf16* __restrict__ yb) {
    size_t idx = (size_t)blockIdx.x * blockDim.x + threadIdx.x;  // B*(SEQ/CHUNK)*D_INNER
    const int c     = (int)(idx % D_INNER);
    const int chunk = (int)((idx / D_INNER) % (SEQ / CHUNK));
    const int b     = (int)(idx / ((size_t)D_INNER * (SEQ / CHUNK)));
    const int s0 = chunk * CHUNK;
    const int sw = (s0 - LOOK > 0) ? (s0 - LOOK) : 0;
    float st = 0.f;
    for (int s = sw; s < s0; ++s)
        st = 0.9f * st + 0.1f * g[((size_t)b * SEQ + s) * D_INNER + c];
    for (int s = s0; s < s0 + CHUNK; ++s) {
        st = 0.9f * st + 0.1f * g[((size_t)b * SEQ + s) * D_INNER + c];
        yb[((size_t)b * SEQ + s) * D_INNER + c] = (__bf16)st;
    }
}

// ---------------------------------------------------------------------------
extern "C" void kernel_launch(void* const* d_in, const int* in_sizes, int n_in,
                              void* d_out, int out_size, void* d_ws, size_t ws_size,
                              hipStream_t stream) {
    (void)in_sizes; (void)n_in; (void)out_size; (void)ws_size;

    const float* x         = (const float*)d_in[0];  // (4,4096,1024)
    const float* in_proj_w = (const float*)d_in[1];  // (3072,1024)
    const float* conv_w    = (const float*)d_in[2];  // (2048,1,4)
    const float* conv_b    = (const float*)d_in[3];  // (2048,)
    const float* dt_proj_w = (const float*)d_in[5];  // (2048,2048)
    const float* dt_proj_b = (const float*)d_in[6];  // (2048,)
    const float* out_proj_w= (const float*)d_in[9];  // (1024,2048)
    float* out = (float*)d_out;                      // (4,4096,1024)

    // Workspace layout (~388 MB)
    char* ws = (char*)d_ws;
    __bf16* xb   = (__bf16*)ws;                ws += (size_t)MTOT * D_MODEL * 2;          // 32 MB
    __bf16* w1b  = (__bf16*)ws;                ws += (size_t)(D_INNER + D_MODEL) * D_MODEL * 2; // 6 MB
    __bf16* w2b  = (__bf16*)ws;                ws += (size_t)D_INNER * D_INNER * 2;       // 8 MB
    __bf16* w3b  = (__bf16*)ws;                ws += (size_t)D_MODEL * D_INNER * 2;       // 4 MB
    float*  u    = (float*)ws;                 ws += (size_t)MTOT * D_INNER * 4;          // 128 MB (u -> gated)
    float*  res  = (float*)ws;                 ws += (size_t)MTOT * D_MODEL * 4;          // 64 MB
    __bf16* ucb  = (__bf16*)ws;                ws += (size_t)MTOT * D_INNER * 2;          // 64 MB
    __bf16* yb   = (__bf16*)ws;                ws += (size_t)MTOT * D_INNER * 2;          // 64 MB

    // 1) Conversions to bf16
    {
        size_t n;
        n = (size_t)MTOT * D_MODEL;
        f32_to_bf16<<<dim3((n + 255) / 256), dim3(256), 0, stream>>>(x, xb, n);
        n = (size_t)(D_INNER + D_MODEL) * D_MODEL;
        f32_to_bf16<<<dim3((n + 255) / 256), dim3(256), 0, stream>>>(in_proj_w, w1b, n);
        n = (size_t)D_INNER * D_INNER;
        f32_to_bf16<<<dim3((n + 255) / 256), dim3(256), 0, stream>>>(dt_proj_w, w2b, n);
        n = (size_t)D_MODEL * D_INNER;
        f32_to_bf16<<<dim3((n + 255) / 256), dim3(256), 0, stream>>>(out_proj_w, w3b, n);
    }

    // 2) in_proj GEMM: 16384 x 3072 x 1024, split into u / res
    gemm_wmma<0><<<dim3((D_INNER + D_MODEL) / 128, MTOT / 128), dim3(256), 0, stream>>>(
        xb, w1b, MTOT, D_INNER + D_MODEL, D_MODEL, u, res, nullptr, nullptr);

    // 3) depthwise causal conv -> ucb (bf16)
    {
        size_t n = (size_t)MTOT * D_INNER;
        dwconv_kernel<<<dim3((unsigned)(n / 256)), dim3(256), 0, stream>>>(u, conv_w, conv_b, ucb);
    }

    // 4) dt_proj GEMM (16384 x 2048 x 2048) with fused gating -> gated (reuses u)
    gemm_wmma<1><<<dim3(D_INNER / 128, MTOT / 128), dim3(256), 0, stream>>>(
        ucb, w2b, MTOT, D_INNER, D_INNER, u, nullptr, dt_proj_b, ucb);

    // 5) chunked decay scan -> yb (bf16)
    {
        size_t n = (size_t)BATCH * (SEQ / CHUNK) * D_INNER;  // 65536
        scan_kernel<<<dim3((unsigned)(n / 256)), dim3(256), 0, stream>>>(u, yb);
    }

    // 6) out_proj GEMM (16384 x 1024 x 2048) with fused +res -> out
    gemm_wmma<2><<<dim3(D_MODEL / 128, MTOT / 128), dim3(256), 0, stream>>>(
        yb, w3b, MTOT, D_MODEL, D_INNER, out, res, nullptr, nullptr);
}
